// MiniMaxMoELayerReference_10840497455872
// MI455X (gfx1250) — matrix operand order repaired and checked by reading, not compiled, not run on hardware
//
#include <hip/hip_runtime.h>
#include <math.h>

// Problem constants (from reference)
#define T_TOK 4096      // B*S
#define DD    2048
#define FF    1024
#define EE    64
#define TOPK  8

typedef __attribute__((ext_vector_type(16))) _Float16 v16h;
typedef __attribute__((ext_vector_type(8)))  _Float16 v8h;
typedef __attribute__((ext_vector_type(8)))  float    v8f;

union AFrag { v16h v; v8h h[2]; };

__device__ __forceinline__ v8f wmma16(v16h a, v16h b, v8f c) {
    // D = A(16x32 f16) * B(32x16 f16) + C(16x16 f32)
    return __builtin_amdgcn_wmma_f32_16x16x32_f16(false, a, false, b,
                                                  (short)0, c, false, false);
}

// B fragment for 16x16x32 f16 WMMA from row-major [K, ldw] fp32 weights.
// lane L<16 : elems 0..7 = W[k0+i][n0+L],  elems 8..15 = W[k0+16+i-8][n0+L]
// lane L>=16: same with +8 row offset, column n0+L-16
__device__ __forceinline__ v16h load_b_frag(const float* __restrict__ W,
                                            int ldw, int k0, int n0, int lane) {
    const int col = n0 + (lane & 15);
    const int kk  = k0 + ((lane >> 4) << 3);
    const float* p = W + (size_t)kk * (size_t)ldw + col;
    v16h b;
#pragma unroll
    for (int i = 0; i < 16; ++i)
        b[i] = (_Float16)p[(size_t)(i + ((i >> 3) << 3)) * (size_t)ldw];
    return b;
}

// A fragment for 16x32 f16 from LDS (row-major, stride in halves, stride%8==0).
// lane L<16 : elems 0..7 = X[row][k0..k0+7],  elems 8..15 = X[row][k0+16..k0+23]
// lane L>=16: same with +8 column offset
__device__ __forceinline__ v16h load_a_frag(const _Float16* base, int stride,
                                            int rowbase, int k0, int lane) {
    const int row  = rowbase + (lane & 15);
    const int koff = (lane >> 4) << 3;
    const _Float16* p = base + row * stride + k0 + koff;
    AFrag a;
    a.h[0] = *(const v8h*)(p);
    a.h[1] = *(const v8h*)(p + 16);
    return a.v;
}

// ---------------------------------------------------------------------------
// Kernel 1: gating — logits, sigmoid, top-8, scatter token lists per expert
// grid: T_TOK/8 blocks of 64 threads (thread = expert)
// ---------------------------------------------------------------------------
__global__ void __launch_bounds__(64)
moe_gate_kernel(const float* __restrict__ x, const float* __restrict__ gw,
                const float* __restrict__ ebias,
                int* __restrict__ cnt, int* __restrict__ tok_idx,
                float* __restrict__ tok_w) {
    __shared__ float xs[8][DD];
    __shared__ float sc[8][EE];
    const int tid = threadIdx.x;
    const int t0  = blockIdx.x * 8;

    for (int i = tid; i < 8 * DD; i += 64) {
        int r = i >> 11, c = i & (DD - 1);
        xs[r][c] = x[(size_t)(t0 + r) * DD + c];
    }
    __syncthreads();

    const float* w = gw + (size_t)tid * DD;
    float acc[8];
#pragma unroll
    for (int t = 0; t < 8; ++t) acc[t] = 0.f;
    for (int k = 0; k < DD; ++k) {
        float wv = w[k];
#pragma unroll
        for (int t = 0; t < 8; ++t) acc[t] = fmaf(wv, xs[t][k], acc[t]);
    }
#pragma unroll
    for (int t = 0; t < 8; ++t)
        sc[t][tid] = 1.f / (1.f + expf(-acc[t]));
    __syncthreads();

    if (tid < 8) {
        const int t = tid;
        unsigned long long chosen = 0ull;
        int   sel[TOPK];
        float sw[TOPK];
        float ssum = 0.f;
#pragma unroll
        for (int i = 0; i < TOPK; ++i) {
            float best = -INFINITY; int bi = 0;
            for (int e = 0; e < EE; ++e) {
                if ((chosen >> e) & 1ull) continue;
                float v = sc[t][e] + ebias[e];
                if (v > best) { best = v; bi = e; }   // first-index tie-break
            }
            chosen |= 1ull << bi;
            sel[i] = bi;
            sw[i]  = sc[t][bi];
            ssum  += sc[t][bi];
        }
        const float inv = 1.0f / (ssum + 1e-20f);     // ROUTED_SCALING = 1.0
#pragma unroll
        for (int i = 0; i < TOPK; ++i) {
            int pos = atomicAdd(&cnt[sel[i]], 1);
            tok_idx[sel[i] * T_TOK + pos] = t0 + t;
            tok_w [sel[i] * T_TOK + pos] = sw[i] * inv;
        }
    }
}

// ---------------------------------------------------------------------------
// Kernel 2: fused expert SwiGLU + down-proj, WMMA f16, atomic scatter-add.
// grid: (T_TOK/64, EE), 256 threads (8 waves). M-tile = 64 gathered tokens.
// ---------------------------------------------------------------------------
#define XS_STRIDE 72     // halves, KC=64 + pad (conflict-free b128)
#define HS_STRIDE 1032   // halves, FF   + pad (conflict-free b128)

__global__ void __launch_bounds__(256)
moe_expert_kernel(const float* __restrict__ x,
                  const float* __restrict__ Wg,
                  const float* __restrict__ Wu,
                  const float* __restrict__ Wd,
                  const int*   __restrict__ cnt,
                  const int*   __restrict__ tok_idx,
                  const float* __restrict__ tok_w,
                  float* __restrict__ out) {
    const int e = blockIdx.y;
    const int n = cnt[e];
    const int row0 = blockIdx.x * 64;
    if (row0 >= n) return;

    __shared__ __align__(16) _Float16 Xs[64 * XS_STRIDE];  // ~9 KB
    __shared__ __align__(16) _Float16 Hs[64 * HS_STRIDE];  // ~129 KB
    __shared__ int   toks[64];
    __shared__ float cws[64];

    const int tid  = threadIdx.x;
    const int lane = tid & 31;
    const int wv   = tid >> 5;

    const float* Wg_e = Wg + (size_t)e * DD * FF;
    const float* Wu_e = Wu + (size_t)e * DD * FF;
    const float* Wd_e = Wd + (size_t)e * FF * DD;

    if (tid < 64) {
        int g  = row0 + tid;
        int gi = (g < n) ? g : (n - 1);
        toks[tid] = tok_idx[e * T_TOK + gi];
        cws[tid]  = (g < n) ? tok_w[e * T_TOK + gi] : 0.f;
    }
    __syncthreads();

    // ---------------- Phase 1: hg = X@Wg, hu = X@Wu, h = c*silu(hg)*hu -----
    // Each wave: 8 N-pair tiles total, 2 per subpass, all 4 row-tiles shared
    for (int sp = 0; sp < 4; ++sp) {
        v8f ag[2][4], au[2][4];
#pragma unroll
        for (int j = 0; j < 2; ++j)
#pragma unroll
            for (int r = 0; r < 4; ++r) { ag[j][r] = (v8f)(0.f); au[j][r] = (v8f)(0.f); }

        const int np0 = wv * 8 + sp * 2;

        for (int kc = 0; kc < DD; kc += 64) {
            __syncthreads();
            // stage X chunk [64 rows][64 cols] as f16 (gathered rows)
            for (int i = tid; i < 64 * 64; i += 256) {
                int r = i >> 6, c = i & 63;
                Xs[r * XS_STRIDE + c] =
                    (_Float16)x[(size_t)toks[r] * DD + kc + c];
            }
            __syncthreads();

#pragma unroll
            for (int k2 = 0; k2 < 64; k2 += 32) {
                v16h a[4];
#pragma unroll
                for (int r = 0; r < 4; ++r)
                    a[r] = load_a_frag(Xs, XS_STRIDE, r * 16, k2, lane);
#pragma unroll
                for (int j = 0; j < 2; ++j) {
                    const int n0 = (np0 + j) * 16;
                    v16h bg = load_b_frag(Wg_e, FF, kc + k2, n0, lane);
#pragma unroll
                    for (int r = 0; r < 4; ++r) ag[j][r] = wmma16(a[r], bg, ag[j][r]);
                    v16h bu = load_b_frag(Wu_e, FF, kc + k2, n0, lane);
#pragma unroll
                    for (int r = 0; r < 4; ++r) au[j][r] = wmma16(a[r], bu, au[j][r]);
                }
            }
        }

        // h = c * silu(g) * u  -> LDS (f16)
#pragma unroll
        for (int j = 0; j < 2; ++j) {
            const int nc = (np0 + j) * 16 + (lane & 15);
#pragma unroll
            for (int r4 = 0; r4 < 4; ++r4) {
#pragma unroll
                for (int i = 0; i < 8; ++i) {
                    const int row = r4 * 16 + i + ((lane >> 4) << 3);
                    float g = ag[j][r4][i];
                    float u = au[j][r4][i];
                    float hv = cws[row] * (g / (1.f + expf(-g))) * u;
                    Hs[row * HS_STRIDE + nc] = (_Float16)hv;
                }
            }
        }
    }
    __syncthreads();

    // ---------------- Phase 2: out += h @ Wd (scatter, atomic) -------------
    // 128 N-tiles of 16 over D; each wave: 16 tiles, 2 per subpass, 4 rts
    for (int sp = 0; sp < 8; ++sp) {
        v8f acc[2][4];
#pragma unroll
        for (int j = 0; j < 2; ++j)
#pragma unroll
            for (int r = 0; r < 4; ++r) acc[j][r] = (v8f)(0.f);

        const int nt0 = wv * 16 + sp * 2;

        for (int k0 = 0; k0 < FF; k0 += 32) {
            v16h a[4];
#pragma unroll
            for (int r = 0; r < 4; ++r)
                a[r] = load_a_frag(Hs, HS_STRIDE, r * 16, k0, lane);
#pragma unroll
            for (int j = 0; j < 2; ++j) {
                const int n0 = (nt0 + j) * 16;
                v16h b = load_b_frag(Wd_e, DD, k0, n0, lane);
#pragma unroll
                for (int r = 0; r < 4; ++r) acc[j][r] = wmma16(a[r], b, acc[j][r]);
            }
        }

#pragma unroll
        for (int j = 0; j < 2; ++j) {
            const int col = (nt0 + j) * 16 + (lane & 15);
#pragma unroll
            for (int r4 = 0; r4 < 4; ++r4) {
#pragma unroll
                for (int i = 0; i < 8; ++i) {
                    const int row = r4 * 16 + i + ((lane >> 4) << 3);
                    if (row0 + row < n) {
                        atomicAdd(&out[(size_t)toks[row] * DD + col],
                                  acc[j][r4][i]);
                    }
                }
            }
        }
    }
}

// ---------------------------------------------------------------------------
extern "C" void kernel_launch(void* const* d_in, const int* in_sizes, int n_in,
                              void* d_out, int out_size, void* d_ws, size_t ws_size,
                              hipStream_t stream) {
    const float* x     = (const float*)d_in[0];
    const float* gw    = (const float*)d_in[1];
    const float* ebias = (const float*)d_in[2];
    const float* Wg    = (const float*)d_in[3];
    const float* Wu    = (const float*)d_in[4];
    const float* Wd    = (const float*)d_in[5];
    float* out = (float*)d_out;

    // workspace layout
    int*   cnt     = (int*)d_ws;                 // 64 ints
    int*   tok_idx = cnt + 64;                   // 64*4096 ints
    float* tok_w   = (float*)(tok_idx + EE * T_TOK); // 64*4096 floats

    hipMemsetAsync(out, 0, (size_t)T_TOK * DD * sizeof(float), stream);
    hipMemsetAsync(cnt, 0, EE * sizeof(int), stream);

    moe_gate_kernel<<<T_TOK / 8, 64, 0, stream>>>(x, gw, ebias,
                                                  cnt, tok_idx, tok_w);

    dim3 grid_e(T_TOK / 64, EE);
    moe_expert_kernel<<<grid_e, 256, 0, stream>>>(x, Wg, Wu, Wd,
                                                  cnt, tok_idx, tok_w, out);
}
